// DecoderRNN_76690936037626
// MI455X (gfx1250) — matrix operand order repaired and checked by reading, not compiled, not run on hardware
//
#include <hip/hip_runtime.h>
#include <hip/hip_bf16.h>
#include <math.h>

// Problem constants from the reference
#define V_    32000
#define E_    256
#define H_    512
#define B_    32
#define S_    400
#define T_    64
#define OOV_  50
#define VEXT  (V_ + OOV_)   // 32050
#define NSTEP (T_ - 1)      // 63
#define NT_   (V_ / 16)     // 2000 N-tiles in the vocab GEMM

typedef __attribute__((ext_vector_type(16))) __bf16 v16bf;
typedef __attribute__((ext_vector_type(8)))  float  v8f;
typedef unsigned short ushort_t;

// ---------- helpers ----------
__device__ __forceinline__ unsigned short f2bf(float f) {
  union { float f; unsigned u; } v; v.f = f;
  unsigned r = v.u + 0x7FFFu + ((v.u >> 16) & 1u);  // round-to-nearest-even
  return (unsigned short)(r >> 16);
}
__device__ __forceinline__ float sigmoidf_(float x) { return 1.0f / (1.0f + expf(-x)); }
__device__ __forceinline__ float wave_sum(float v) {
  #pragma unroll
  for (int o = 16; o > 0; o >>= 1) v += __shfl_xor(v, o, 32);
  return v;
}
__device__ __forceinline__ float wave_max(float v) {
  #pragma unroll
  for (int o = 16; o > 0; o >>= 1) v = fmaxf(v, __shfl_xor(v, o, 32));
  return v;
}

// ---------- prep kernels ----------
// h := encoder_hidden[0]; last_out := 0
__global__ void k_init(const float* __restrict__ enc_h, float* __restrict__ h,
                       float* __restrict__ co) {
  int i = blockIdx.x * 256 + threadIdx.x;
  if (i < B_ * H_) { h[i] = enc_h[i]; co[i] = 0.0f; }
}

// Pack W_out (V x H, row-major f32) into bf16 B-fragments:
// Bpack[nt][kb][lane][16] ; lane%16 = N, K = kb*32 + (lane/16)*16 + i (contiguous)
__global__ void k_pack_wout(const float* __restrict__ W_out, ushort_t* __restrict__ bp) {
  int tid = blockIdx.x * 256 + threadIdx.x;           // 2000*16*32 threads
  if (tid >= NT_ * 16 * 32) return;
  int lane = tid & 31;
  int kb   = (tid >> 5) & 15;
  int nt   = tid >> 9;
  int n     = nt * 16 + (lane & 15);
  int kbase = kb * 32 + ((lane >> 4) << 4);
  const float* src = W_out + (size_t)n * H_ + kbase;
  ushort_t* dst = bp + ((size_t)(nt * 16 + kb) * 32 + lane) * 16;
  #pragma unroll
  for (int i = 0; i < 16; ++i) dst[i] = f2bf(src[i]);
}

// ---------- per-step kernels ----------
// GRU cell: one wave per output element (b, j). K-parallel over lanes.
__global__ void k_gru(const int* __restrict__ target, int t,
                      const float* __restrict__ emb,
                      const float* __restrict__ h_in,      // B x H  (h)
                      const float* __restrict__ last_out,  // B x H  (co from prev step)
                      const float* __restrict__ W_ih, const float* __restrict__ W_hh,
                      const float* __restrict__ b_ih, const float* __restrict__ b_hh,
                      float* __restrict__ h_out) {
  int wave = blockIdx.x * (blockDim.x >> 5) + (threadIdx.x >> 5);
  int lane = threadIdx.x & 31;
  if (wave >= B_ * H_) return;
  int b = wave >> 9;          // /512
  int j = wave & (H_ - 1);
  int tok = target[b * T_ + t];
  const float* wr = W_ih + (size_t)j * (E_ + H_);
  const float* wz = W_ih + (size_t)(H_ + j) * (E_ + H_);
  const float* wn = W_ih + (size_t)(2 * H_ + j) * (E_ + H_);
  const float* vr = W_hh + (size_t)j * H_;
  const float* vz = W_hh + (size_t)(H_ + j) * H_;
  const float* vn = W_hh + (size_t)(2 * H_ + j) * H_;
  float ir = 0.f, iz = 0.f, in_ = 0.f, hr = 0.f, hz = 0.f, hn = 0.f;
  for (int k = lane; k < E_ + H_; k += 32) {
    float xk = (k < E_) ? emb[(size_t)tok * E_ + k] : last_out[b * H_ + (k - E_)];
    ir += xk * wr[k]; iz += xk * wz[k]; in_ += xk * wn[k];
    if (k < H_) {
      float hk = h_in[b * H_ + k];
      hr += hk * vr[k]; hz += hk * vz[k]; hn += hk * vn[k];
    }
  }
  ir = wave_sum(ir); iz = wave_sum(iz); in_ = wave_sum(in_);
  hr = wave_sum(hr); hz = wave_sum(hz); hn = wave_sum(hn);
  if (lane == 0) {
    float r = sigmoidf_((ir + b_ih[j]) + (hr + b_hh[j]));
    float z = sigmoidf_((iz + b_ih[H_ + j]) + (hz + b_hh[H_ + j]));
    float n = tanhf((in_ + b_ih[2 * H_ + j]) + r * (hn + b_hh[2 * H_ + j]));
    h_out[b * H_ + j] = (1.0f - z) * n + z * h_in[b * H_ + j];
  }
}

// Attention: q = h_new @ W_attn^T, masked softmax scores, ctx, p_gen. One block per batch row.
__global__ void k_attn(const float* __restrict__ h_new, const float* __restrict__ enc,
                       const float* __restrict__ W_attn, const int* __restrict__ input_lens,
                       const float* __restrict__ emb, const int* __restrict__ target, int t,
                       const float* __restrict__ W_ptr, const float* __restrict__ b_ptr,
                       float* __restrict__ ctx, float* __restrict__ attn,
                       float* __restrict__ pgen) {
  __shared__ float hvec[H_];
  __shared__ float qv[H_];
  __shared__ float sc[S_];
  __shared__ float red[32];
  int b = blockIdx.x;
  int tid = threadIdx.x;              // 512 threads = 16 waves
  int lane = tid & 31, wave = tid >> 5;
  hvec[tid] = h_new[b * H_ + tid];
  __syncthreads();
  // q_j = h_new[b,:] . W_attn[j,:]
  const float* wrow = W_attn + (size_t)tid * H_;
  float acc = 0.f;
  for (int k = 0; k < H_; ++k) acc += hvec[k] * wrow[k];
  qv[tid] = acc;
  __syncthreads();
  // scores (wave per source position, K-parallel over lanes)
  int len = input_lens[b];
  for (int s = wave; s < S_; s += 16) {
    const float* erow = enc + ((size_t)b * S_ + s) * H_;
    float p = 0.f;
    for (int k = lane; k < H_; k += 32) p += qv[k] * erow[k];
    p = wave_sum(p);
    if (lane == 0) sc[s] = (s < len) ? p : -1e9f;
  }
  __syncthreads();
  // masked softmax over S
  float m = -1e30f;
  for (int s = tid; s < S_; s += 512) m = fmaxf(m, sc[s]);
  m = wave_max(m);
  if (lane == 0) red[wave] = m;
  __syncthreads();
  if (tid == 0) { float mm = -1e30f; for (int i = 0; i < 16; ++i) mm = fmaxf(mm, red[i]); red[16] = mm; }
  __syncthreads();
  float rmax = red[16];
  float ssum = 0.f;
  for (int s = tid; s < S_; s += 512) { float e = expf(sc[s] - rmax); sc[s] = e; ssum += e; }
  ssum = wave_sum(ssum);
  __syncthreads();
  if (lane == 0) red[wave] = ssum;
  __syncthreads();
  if (tid == 0) { float tt = 0.f; for (int i = 0; i < 16; ++i) tt += red[i]; red[17] = 1.0f / tt; }
  __syncthreads();
  float inv = red[17];
  for (int s = tid; s < S_; s += 512) { float a = sc[s] * inv; sc[s] = a; attn[b * S_ + s] = a; }
  __syncthreads();
  // ctx_j = sum_s attn[s] * enc[b,s,j]   (coalesced over j)
  float c = 0.f;
  for (int s = 0; s < S_; ++s) c += sc[s] * enc[((size_t)b * S_ + s) * H_ + tid];
  ctx[b * H_ + tid] = c;
  __syncthreads();
  // p_gen = sigmoid(W_ptr . [ctx, h_new, e] + b_ptr)
  int tok = target[b * T_ + t];
  float pp = 0.f;
  for (int i = tid; i < 2 * H_ + E_; i += 512) {
    float comp = (i < H_) ? ctx[b * H_ + i]
               : (i < 2 * H_) ? hvec[i - H_]
               : emb[(size_t)tok * E_ + (i - 2 * H_)];
    pp += W_ptr[i] * comp;
  }
  qv[tid] = pp;
  __syncthreads();
  for (int st = 256; st > 0; st >>= 1) { if (tid < st) qv[tid] += qv[tid + st]; __syncthreads(); }
  if (tid == 0) pgen[b] = sigmoidf_(qv[0] + b_ptr[0]);
}

// co = tanh([h_new, ctx] @ W_c^T + b_c) : one wave per (b, j)
__global__ void k_co(const float* __restrict__ h_new, const float* __restrict__ ctx,
                     const float* __restrict__ W_c, const float* __restrict__ b_c,
                     float* __restrict__ co) {
  int wave = blockIdx.x * (blockDim.x >> 5) + (threadIdx.x >> 5);
  int lane = threadIdx.x & 31;
  if (wave >= B_ * H_) return;
  int b = wave >> 9, j = wave & (H_ - 1);
  const float* w = W_c + (size_t)j * (2 * H_);
  float acc = 0.f;
  for (int k = lane; k < 2 * H_; k += 32) {
    float xk = (k < H_) ? h_new[b * H_ + k] : ctx[b * H_ + (k - H_)];
    acc += xk * w[k];
  }
  acc = wave_sum(acc);
  if (lane == 0) co[b * H_ + j] = tanhf(acc + b_c[j]);
}

// Pack co (32x512 f32) into bf16 A-fragments: Apack[mt][kb][lane][16]
// lane%16 = M offset, K = kb*32 + (lane/16)*16 + i
__global__ void k_packA(const float* __restrict__ co, ushort_t* __restrict__ ap) {
  int tid = blockIdx.x * 256 + threadIdx.x;     // 2*16*32 = 1024 threads
  if (tid >= 2 * 16 * 32) return;
  int lane = tid & 31;
  int kb   = (tid >> 5) & 15;
  int mt   = tid >> 9;
  int M = mt * 16 + (lane & 15);
  ushort_t* dst = ap + ((size_t)(mt * 16 + kb) * 32 + lane) * 16;
  #pragma unroll
  for (int i = 0; i < 16; ++i) {
    int k = kb * 32 + ((lane >> 4) << 4) + i;
    dst[i] = f2bf(co[(size_t)M * H_ + k]);
  }
}

// logits = co @ W_out^T + b_out via v_wmma_f32_16x16x32_bf16.
// A-fragments staged in LDS once per block; 8 waves/block, 1 N-tile/wave, 32 WMMA/wave.
__global__ void k_gemm_out(const ushort_t* __restrict__ apack,
                           const ushort_t* __restrict__ bpack,
                           const float* __restrict__ b_out,
                           float* __restrict__ logits) {
  __shared__ __align__(32) ushort_t sA[2 * 16 * 32 * 16];   // 32 KB
  int tid = threadIdx.x;   // 256
  {
    const uint4* src = (const uint4*)apack;
    uint4* dst = (uint4*)sA;
    #pragma unroll
    for (int i = tid; i < (2 * 16 * 32 * 16 * 2) / 16; i += 256) dst[i] = src[i];
  }
  __syncthreads();
  int lane = tid & 31;
  int nt = blockIdx.x * 8 + (tid >> 5);
  v8f c0 = {}; v8f c1 = {};
  #pragma unroll
  for (int kb = 0; kb < 16; ++kb) {
    v16bf bf = *(const v16bf*)(bpack + ((size_t)(nt * 16 + kb) * 32 + lane) * 16);
    v16bf a0 = *(const v16bf*)(sA + ((size_t)(kb)      * 32 + lane) * 16);
    v16bf a1 = *(const v16bf*)(sA + ((size_t)(16 + kb) * 32 + lane) * 16);
    c0 = __builtin_amdgcn_wmma_f32_16x16x32_bf16(false, a0, false, bf, (short)0, c0, false, false);
    c1 = __builtin_amdgcn_wmma_f32_16x16x32_bf16(false, a1, false, bf, (short)0, c1, false, false);
  }
  int n = nt * 16 + (lane & 15);
  float bias = b_out[n];
  int mbase = (lane >= 16) ? 8 : 0;
  #pragma unroll
  for (int v = 0; v < 8; ++v) {
    int m0 = mbase + v;
    logits[(size_t)m0 * V_ + n]        = c0[v] + bias;   // rows 0..15
    logits[(size_t)(m0 + 16) * V_ + n] = c1[v] + bias;   // rows 16..31
  }
}

// Row max + sum(exp) over vocab, one block per batch row.
__global__ void k_sm(const float* __restrict__ logits, float* __restrict__ rmax,
                     float* __restrict__ rsum) {
  __shared__ float red[16];
  int b = blockIdx.x, tid = threadIdx.x;          // 256 threads
  int lane = tid & 31, wave = tid >> 5;
  const float* row = logits + (size_t)b * V_;
  float m = -1e30f;
  for (int v = tid; v < V_; v += 256) m = fmaxf(m, row[v]);
  m = wave_max(m);
  if (lane == 0) red[wave] = m;
  __syncthreads();
  if (tid == 0) { float mm = -1e30f; for (int i = 0; i < 8; ++i) mm = fmaxf(mm, red[i]); red[8] = mm; }
  __syncthreads();
  float bm = red[8];
  float s = 0.f;
  for (int v = tid; v < V_; v += 256) s += expf(row[v] - bm);
  s = wave_sum(s);
  __syncthreads();
  if (lane == 0) red[wave] = s;
  __syncthreads();
  if (tid == 0) {
    float tt = 0.f; for (int i = 0; i < 8; ++i) tt += red[i];
    rmax[b] = bm; rsum[b] = tt;
  }
}

// p_ext[b, v<V] = p_gen * softmax(logits); OOV slots zeroed. Also emit p_gens output.
__global__ void k_pext(const float* __restrict__ logits, const float* __restrict__ rmax,
                       const float* __restrict__ rsum, const float* __restrict__ pgen,
                       float* __restrict__ pext, float* __restrict__ out, int t) {
  int idx = blockIdx.x * 256 + threadIdx.x;
  if (idx >= B_ * VEXT) return;
  int b = idx / VEXT, v = idx - b * VEXT;
  float pg = pgen[b];
  float p = 0.f;
  if (v < V_) p = pg * expf(logits[(size_t)b * V_ + v] - rmax[b]) / rsum[b];
  pext[idx] = p;
  if (v == 0) out[(size_t)NSTEP * B_ * VEXT + (size_t)t * B_ + b] = pg;
}

// Copy-mechanism scatter-add: p_ext[b, inputs_raw[b,s]] += (1-p_gen)*attn[b,s]
__global__ void k_scatter(const int* __restrict__ inputs_raw, const float* __restrict__ attn,
                          const float* __restrict__ pgen, float* __restrict__ pext) {
  int idx = blockIdx.x * 256 + threadIdx.x;
  if (idx >= B_ * S_) return;
  int b = idx / S_, s = idx - b * S_;
  float val = (1.0f - pgen[b]) * attn[b * S_ + s];
  int col = inputs_raw[b * S_ + s];
  atomicAdd(&pext[(size_t)b * VEXT + col], val);
}

__global__ void k_log(const float* __restrict__ pext, float* __restrict__ out, int t) {
  int idx = blockIdx.x * 256 + threadIdx.x;
  if (idx >= B_ * VEXT) return;
  out[(size_t)t * B_ * VEXT + idx] = logf(pext[idx] + 1e-9f);
}

// ---------- workspace layout (bytes, 256-aligned) ----------
#define OFF_WOUT   ((size_t)0)                          // 32,768,000  bf16 W_out fragments
#define OFF_LOGITS (OFF_WOUT   + (size_t)V_ * H_ * 2)   //  4,096,000  f32 logits
#define OFF_PEXT   (OFF_LOGITS + (size_t)B_ * V_ * 4)   //  4,102,400  f32 p_ext
#define OFF_HA     (OFF_PEXT   + (size_t)B_ * VEXT * 4)
#define OFF_HB     (OFF_HA     + (size_t)B_ * H_ * 4)
#define OFF_CO     (OFF_HB     + (size_t)B_ * H_ * 4)
#define OFF_COPACK (OFF_CO     + (size_t)B_ * H_ * 4)   // 32 KB bf16 A fragments
#define OFF_CTX    (OFF_COPACK + (size_t)2 * 16 * 32 * 16 * 2)
#define OFF_ATTN   (OFF_CTX    + (size_t)B_ * H_ * 4)
#define OFF_PGEN   (OFF_ATTN   + (size_t)B_ * S_ * 4)
#define OFF_RMAX   (OFF_PGEN   + 256)
#define OFF_RSUM   (OFF_RMAX   + 256)

extern "C" void kernel_launch(void* const* d_in, const int* in_sizes, int n_in,
                              void* d_out, int out_size, void* d_ws, size_t ws_size,
                              hipStream_t stream) {
  const int*   target     = (const int*)d_in[0];
  const int*   inputs_raw = (const int*)d_in[1];
  const int*   input_lens = (const int*)d_in[2];
  /* d_in[3] = oov_size scalar (constant 50, baked in) */
  const float* enc_h      = (const float*)d_in[4];
  const float* enc_o      = (const float*)d_in[5];
  const float* emb        = (const float*)d_in[6];
  const float* W_ih       = (const float*)d_in[7];
  const float* W_hh       = (const float*)d_in[8];
  const float* b_ih       = (const float*)d_in[9];
  const float* b_hh       = (const float*)d_in[10];
  const float* W_attn     = (const float*)d_in[11];
  const float* W_c        = (const float*)d_in[12];
  const float* b_c        = (const float*)d_in[13];
  const float* W_out      = (const float*)d_in[14];
  const float* b_out      = (const float*)d_in[15];
  const float* W_ptr      = (const float*)d_in[16];
  const float* b_ptr      = (const float*)d_in[17];

  char* ws = (char*)d_ws;
  ushort_t* woutp  = (ushort_t*)(ws + OFF_WOUT);
  float*    logits = (float*)(ws + OFF_LOGITS);
  float*    pext   = (float*)(ws + OFF_PEXT);
  float*    hA     = (float*)(ws + OFF_HA);
  float*    hB     = (float*)(ws + OFF_HB);
  float*    co     = (float*)(ws + OFF_CO);
  ushort_t* copack = (ushort_t*)(ws + OFF_COPACK);
  float*    ctx    = (float*)(ws + OFF_CTX);
  float*    attn   = (float*)(ws + OFF_ATTN);
  float*    pgen   = (float*)(ws + OFF_PGEN);
  float*    rmax   = (float*)(ws + OFF_RMAX);
  float*    rsum   = (float*)(ws + OFF_RSUM);
  float*    out    = (float*)d_out;

  // Prep: init carry state; pack W_out to bf16 fragments (stays hot in 192MB L2)
  k_init<<<(B_ * H_ + 255) / 256, 256, 0, stream>>>(enc_h, hA, co);
  k_pack_wout<<<(NT_ * 16 * 32 + 255) / 256, 256, 0, stream>>>(W_out, woutp);

  float* h_in = hA;
  float* h_out = hB;
  for (int t = 0; t < NSTEP; ++t) {
    k_gru<<<(B_ * H_ + 7) / 8, 256, 0, stream>>>(target, t, emb, h_in, co,
                                                 W_ih, W_hh, b_ih, b_hh, h_out);
    k_attn<<<B_, 512, 0, stream>>>(h_out, enc_o, W_attn, input_lens, emb, target, t,
                                   W_ptr, b_ptr, ctx, attn, pgen);
    k_co<<<(B_ * H_ + 7) / 8, 256, 0, stream>>>(h_out, ctx, W_c, b_c, co);
    k_packA<<<4, 256, 0, stream>>>(co, copack);
    k_gemm_out<<<NT_ / 8, 256, 0, stream>>>(copack, woutp, b_out, logits);
    k_sm<<<B_, 256, 0, stream>>>(logits, rmax, rsum);
    k_pext<<<(B_ * VEXT + 255) / 256, 256, 0, stream>>>(logits, rmax, rsum, pgen,
                                                        pext, out, t);
    k_scatter<<<(B_ * S_ + 255) / 256, 256, 0, stream>>>(inputs_raw, attn, pgen, pext);
    k_log<<<(B_ * VEXT + 255) / 256, 256, 0, stream>>>(pext, out, t);
    float* tmp = h_in; h_in = h_out; h_out = tmp;   // ping-pong hidden state
  }
}
